// RickerCWT_63797444215779
// MI455X (gfx1250) — matrix operand order repaired and checked by reading, not compiled, not run on hardware
//
#include <hip/hip_runtime.h>

// Ricker CWT as implicit GEMM on CDNA5 WMMA (bf16x3 split for ~f32 accuracy).
// out[b,s,t] = sum_k x[b, t+k-271] * W[s,k];  B=128, T=4096, S=64, K=543.

typedef unsigned int u32;
typedef __attribute__((ext_vector_type(16))) __bf16 bf16x16;
typedef __attribute__((ext_vector_type(8)))  float   f32x8;
typedef __attribute__((ext_vector_type(8)))  u32     u32x8;

#define NB     128
#define NT     4096
#define NS     64
#define KTAPS  543
#define KPAD   544      // taps padded to multiple of 32 (tap 543 == 0)
#define HALF_K 271
#define WPITCH 272      // bf16-pair dwords per scale row (KPAD/2)
#define TTILE  256      // time positions per workgroup
#define STRIP  800      // TTILE + KPAD input samples staged in LDS
#define OPITCH 260      // padded LDS output pitch (260 % 64 == 4 -> no bank conflicts)

static __device__ __forceinline__ u32 bf16_rne(float f) {
    u32 u = __builtin_bit_cast(u32, f);
    return (u + 0x7FFFu + ((u >> 16) & 1u)) >> 16;   // round-to-nearest-even
}
static __device__ __forceinline__ float bf16_to_f32(u32 h) {
    return __builtin_bit_cast(float, h << 16);
}
// pack {hi(x) in [15:0], lo(x) in [31:16]} for one f32 sample
static __device__ __forceinline__ u32 split_pack(float x) {
    u32 h = bf16_rne(x);
    u32 l = bf16_rne(x - bf16_to_f32(h));
    return h | (l << 16);
}

// ---------------------------------------------------------------------------
// Prep: W [64][543] f32  ->  whi/wlo [64][272] dwords, each dword = bf16 pair
// {tap 2j, tap 2j+1} so a B fragment (16x16x32 bf16, lane n = scale column,
// K linear) is two aligned 16B loads per lane.
// ---------------------------------------------------------------------------
__global__ __launch_bounds__(256) void ricker_prep(const float* __restrict__ W,
                                                   u32* __restrict__ whi,
                                                   u32* __restrict__ wlo) {
    int i = blockIdx.x * 256 + threadIdx.x;
    if (i >= NS * WPITCH) return;
    int s = i / WPITCH;
    int j = i - s * WPITCH;
    int t0 = 2 * j;
    float w0 = W[s * KTAPS + t0];
    float w1 = (t0 + 1 < KTAPS) ? W[s * KTAPS + t0 + 1] : 0.0f;
    u32 h0 = bf16_rne(w0);
    u32 l0 = bf16_rne(w0 - bf16_to_f32(h0));
    u32 h1 = bf16_rne(w1);
    u32 l1 = bf16_rne(w1 - bf16_to_f32(h1));
    whi[i] = h0 | (h1 << 16);
    wlo[i] = l0 | (l1 << 16);
}

// ---------------------------------------------------------------------------
// Main: one workgroup = (batch b, 256-time tile). 8 waves; wave w computes
// 32 time rows x 64 scales via 2x4 accumulator tiles, K-loop over 17 chunks
// of 32 taps, 3 bf16 WMMAs per (acc tile, chunk).
// ---------------------------------------------------------------------------
__global__ __launch_bounds__(256) void ricker_cwt(const float* __restrict__ x,
                                                  const u32* __restrict__ whi,
                                                  const u32* __restrict__ wlo,
                                                  float* __restrict__ out) {
    __shared__ u32   xstrip[STRIP];        // packed {hi,lo} bf16 per sample
    __shared__ float obuf[NS * OPITCH];    // [scale][time] staging for coalesced WB

    const int tid = threadIdx.x;
    const int b   = blockIdx.y;
    const int t0  = blockIdx.x * TTILE;

    // Stage zero-padded input strip: strip[i] <-> x[b][t0 + i - 271]
    for (int i = tid; i < STRIP; i += 256) {
        int g = t0 + i - HALF_K;
        float v = (g >= 0 && g < NT) ? x[b * NT + g] : 0.0f;
        xstrip[i] = split_pack(v);
    }
    __syncthreads();

    const int lane = tid & 31;
    const int wv   = tid >> 5;
    const int lm   = lane & 15;
    const int h    = lane >> 4;   // lane half select
    const int m0   = wv * 32;     // wave's time base within tile

    f32x8 acc[2][4];
    #pragma unroll
    for (int mt = 0; mt < 2; ++mt)
        #pragma unroll
        for (int nt = 0; nt < 4; ++nt)
            acc[mt][nt] = (f32x8)0.0f;

    for (int kc = 0; kc < KPAD / 32; ++kc) {
        // A fragments (Hankel): this code's k order (B matches) = lanes 0-15 hold K {0..15},
        // lanes 16-31 hold K {16..31}; dword j = taps {2j,2j+1}.
        u32x8 ahi[2], alo[2];
        #pragma unroll
        for (int mt = 0; mt < 2; ++mt) {
            int p0 = m0 + mt * 16 + lm + kc * 32 + h * 16;
            u32 e[16];
            #pragma unroll
            for (int c = 0; c < 8; ++c) {
                e[c]     = xstrip[p0 + c];
                e[8 + c] = xstrip[p0 + 8 + c];
            }
            #pragma unroll
            for (int j = 0; j < 8; ++j) {
                u32 a = e[2 * j], c2 = e[2 * j + 1];
                ahi[mt][j] = (a & 0xFFFFu) | (c2 << 16);
                alo[mt][j] = (a >> 16)     | (c2 & 0xFFFF0000u);
            }
        }
        #pragma unroll
        for (int nt = 0; nt < 4; ++nt) {
            int s    = nt * 16 + lm;
            int bidx = s * WPITCH + kc * 16 + h * 8;          // 32B aligned
            u32x8 bh = *(const u32x8*)(whi + bidx);           // 2x global_load_b128
            u32x8 bl = *(const u32x8*)(wlo + bidx);
            bf16x16 bhf = __builtin_bit_cast(bf16x16, bh);
            bf16x16 blf = __builtin_bit_cast(bf16x16, bl);
            #pragma unroll
            for (int mt = 0; mt < 2; ++mt) {
                bf16x16 ahf = __builtin_bit_cast(bf16x16, ahi[mt]);
                bf16x16 alf = __builtin_bit_cast(bf16x16, alo[mt]);
                acc[mt][nt] = __builtin_amdgcn_wmma_f32_16x16x32_bf16(
                    false, ahf, false, bhf, (short)0, acc[mt][nt], false, false);
                acc[mt][nt] = __builtin_amdgcn_wmma_f32_16x16x32_bf16(
                    false, ahf, false, blf, (short)0, acc[mt][nt], false, false);
                acc[mt][nt] = __builtin_amdgcn_wmma_f32_16x16x32_bf16(
                    false, alf, false, bhf, (short)0, acc[mt][nt], false, false);
            }
        }
    }

    // C layout: dword j, lanes 0-15 -> M=j, N=lane; lanes 16-31 -> M=8+j.
    // M = time offset, N = scale offset. Stage into padded LDS tile.
    #pragma unroll
    for (int mt = 0; mt < 2; ++mt) {
        #pragma unroll
        for (int nt = 0; nt < 4; ++nt) {
            int s  = nt * 16 + lm;
            int tt = m0 + mt * 16 + h * 8;
            #pragma unroll
            for (int j = 0; j < 8; j += 2) {
                float2 v;
                v.x = acc[mt][nt][j];
                v.y = acc[mt][nt][j + 1];
                *(float2*)&obuf[s * OPITCH + tt + j] = v;     // ds_store_b64
            }
        }
    }
    __syncthreads();

    // Coalesced writeback: 64 scales x 256 times, float4 per thread per row-group.
    const int col = (tid & 63) * 4;
    const int rof = tid >> 6;
    #pragma unroll
    for (int it = 0; it < 16; ++it) {
        int r = it * 4 + rof;
        float4 v = *(const float4*)&obuf[r * OPITCH + col];
        *(float4*)(out + ((size_t)(b * NS + r)) * NT + t0 + col) = v;
    }
}

extern "C" void kernel_launch(void* const* d_in, const int* in_sizes, int n_in,
                              void* d_out, int out_size, void* d_ws, size_t ws_size,
                              hipStream_t stream) {
    const float* x = (const float*)d_in[0];   // [128, 4096] f32
    const float* W = (const float*)d_in[1];   // [64, 543]   f32
    u32* whi = (u32*)d_ws;                    // [64][272] bf16-pair dwords
    u32* wlo = whi + NS * WPITCH;             // ws usage: 139,264 bytes total

    ricker_prep<<<(NS * WPITCH + 255) / 256, 256, 0, stream>>>(W, whi, wlo);

    dim3 grid(NT / TTILE, NB);                // (16, 128) workgroups
    ricker_cwt<<<grid, dim3(256), 0, stream>>>(x, whi, wlo, (float*)d_out);
}
